// INNModel_52819507806750
// MI455X (gfx1250) — matrix operand order repaired and checked by reading, not compiled, not run on hardware
//
#include <hip/hip_runtime.h>
#include <hip/hip_bf16.h>

// ---------------- problem constants (match reference) ----------------
#define VV   50000
#define NELC 200
#define DD   256
#define RR   256
#define HIDC 256
#define H2C  512           // 2*HID
#define LL   2048
#define NEC  512
#define KK   8
#define NSC  1024
#define G4   1024          // 4*HID
#define ZDIM 1280          // R + 2*H2
#define IOU3 1536          // 3*H2

typedef __attribute__((ext_vector_type(16))) _Float16 v16h;
typedef __attribute__((ext_vector_type(8)))  _Float16 v8h;
typedef __attribute__((ext_vector_type(8)))  float    v8f;

__device__ __forceinline__ float sigf(float x) { return 1.0f / (1.0f + expf(-x)); }

// ---------------- K0: embedding gather + f16 convert ----------------
__global__ void embed_f16_kernel(const int* __restrict__ tokens,
                                 const float* __restrict__ wemb,
                                 _Float16* __restrict__ x16) {
    int i = blockIdx.x * blockDim.x + threadIdx.x;     // 0 .. L*D-1
    int l = i >> 8;                                    // / D (D==256)
    int d = i & 255;
    x16[i] = (_Float16)wemb[tokens[l] * DD + d];
}

// ---------------- K0b: weight transpose + f16 convert: WT[n*K+k] = W[k*N+n] -------
__global__ void wtrans_f16_kernel(const float* __restrict__ W,
                                  _Float16* __restrict__ WT,
                                  int Kd, int N) {
    int i = blockIdx.x * blockDim.x + threadIdx.x;     // 0 .. K*N-1, coalesced read
    int k = i / N, n = i - k * N;
    WT[(size_t)n * Kd + k] = (_Float16)W[i];
}

// ---------------- K1: WMMA GEMM  C[M,N] = A_f16[M,Kd] @ BT_f16[N,Kd]^T ------------
// One wave computes a 16x64 output strip: 1 shared A fragment, 4 B fragments,
// 4 x v_wmma_f32_16x16x32_f16 per k-step. All fragment loads are 128-bit.
__global__ void gemm_wmma_kernel(const _Float16* __restrict__ A,
                                 const _Float16* __restrict__ BT,  // N x Kd (pre-transposed)
                                 float* __restrict__ C,
                                 int M, int N, int Kd) {
    int wave = (blockIdx.x * blockDim.x + threadIdx.x) >> 5;
    int lane = threadIdx.x & 31;
    int ngrp = N >> 6;                                 // 64-wide column groups
    int mt = wave / ngrp;
    int ng = wave - mt * ngrp;
    if (mt >= (M >> 4)) return;                        // wave-uniform: EXEC stays all-1

    int m   = lane & 15;                               // A row / B,C column (within tile)
    int hlf = lane >> 4;
    int n0  = ng << 6;

    v8f c0 = {}, c1 = {}, c2 = {}, c3 = {};
    const _Float16* Arow = A + (size_t)(mt * 16 + m) * Kd;
    // B fragment base for tile j: column n0+16j+m, 16 contiguous halfs at K=kk+16*hlf
    const _Float16* B0 = BT + (size_t)(n0 + 0  + m) * Kd + 16 * hlf;
    const _Float16* B1 = BT + (size_t)(n0 + 16 + m) * Kd + 16 * hlf;
    const _Float16* B2 = BT + (size_t)(n0 + 32 + m) * Kd + 16 * hlf;
    const _Float16* B3 = BT + (size_t)(n0 + 48 + m) * Kd + 16 * hlf;

    for (int kk = 0; kk < Kd; kk += 32) {
        // A 16x32 f16 fragment: element e -> K=kk+16*hlf+e (contiguous, matches B).
        v16h a = *(const v16h*)(Arow + kk + 16 * hlf);
        // B 32x16 fragment: element e -> K=kk+16*hlf+e, column fixed -> contiguous.
        v16h b0 = *(const v16h*)(B0 + kk);
        v16h b1 = *(const v16h*)(B1 + kk);
        v16h b2 = *(const v16h*)(B2 + kk);
        v16h b3 = *(const v16h*)(B3 + kk);
        c0 = __builtin_amdgcn_wmma_f32_16x16x32_f16(false, a, false, b0, (short)0, c0, false, false);
        c1 = __builtin_amdgcn_wmma_f32_16x16x32_f16(false, a, false, b1, (short)0, c1, false, false);
        c2 = __builtin_amdgcn_wmma_f32_16x16x32_f16(false, a, false, b2, (short)0, c2, false, false);
        c3 = __builtin_amdgcn_wmma_f32_16x16x32_f16(false, a, false, b3, (short)0, c3, false, false);
    }
    // C/D 16x16 f32 layout: lane l, VGPR r -> (row = r + 8*hlf, col = lane&15)
    #pragma unroll
    for (int r = 0; r < 8; ++r) {
        size_t row = (size_t)(mt * 16 + r + 8 * hlf) * N + n0 + m;
        C[row +  0] = c0[r];
        C[row + 16] = c1[r];
        C[row + 32] = c2[r];
        C[row + 48] = c3[r];
    }
}

// ---------------- K2: sequential LSTM scan (block 0 = fwd, block 1 = bwd) ----------
// 1024 threads: each thread owns one of the 4*HID gate columns for h@Whh.
__global__ void lstm_scan_kernel(const float* __restrict__ xgF,
                                 const float* __restrict__ xgB,
                                 const float* __restrict__ WhhF,
                                 const float* __restrict__ WhhB,
                                 const float* __restrict__ bF,
                                 const float* __restrict__ bB,
                                 float* __restrict__ blstm) {
    __shared__ float h[HIDC];
    __shared__ float cbuf[HIDC];
    __shared__ float g[G4];

    int dir = blockIdx.x;
    const float* xg   = dir ? xgB  : xgF;
    const float* Whh  = dir ? WhhB : WhhF;
    const float* bias = dir ? bB   : bF;
    int tid = threadIdx.x;                             // 0..1023

    if (tid < HIDC) { h[tid] = 0.0f; cbuf[tid] = 0.0f; }
    __syncthreads();

    const float* wcol = Whh + tid;                     // column `tid`, stride G4
    __builtin_prefetch(wcol, 0, 1);                    // global_prefetch_b8

    for (int s = 0; s < LL; ++s) {
        int t = dir ? (LL - 1 - s) : s;
        float acc = xg[(size_t)t * G4 + tid] + bias[tid];
        #pragma unroll 8
        for (int k = 0; k < HIDC; ++k)
            acc = fmaf(h[k], wcol[(size_t)k * G4], acc);
        g[tid] = acc;
        __syncthreads();
        if (tid < HIDC) {
            float ig = g[tid], fg = g[HIDC + tid];
            float gg = g[2 * HIDC + tid], og = g[3 * HIDC + tid];
            float c = sigf(fg) * cbuf[tid] + sigf(ig) * tanhf(gg);
            float hh = sigf(og) * tanhf(c);
            cbuf[tid] = c;
            h[tid] = hh;
            blstm[(size_t)t * H2C + dir * HIDC + tid] = hh;
        }
        __syncthreads();
    }
}

// ---------------- K3: attention scores  scores[l] = blstm[l] . Wa + ba ------------
__global__ void score_kernel(const float* __restrict__ blstm,
                             const float* __restrict__ Wa,
                             const float* __restrict__ ba,
                             float* __restrict__ scores) {
    int l = blockIdx.x * blockDim.x + threadIdx.x;
    if (l >= LL) return;
    float acc = ba[0];
    const float* row = blstm + (size_t)l * H2C;
    for (int d = 0; d < H2C; ++d) acc = fmaf(row[d], Wa[d], acc);
    scores[l] = acc;
}

// ---------------- K4: per-entity softmax pooling -> Hbuf[0:NE] --------------------
__global__ void entity_pool_kernel(const int* __restrict__ spans,
                                   const float* __restrict__ scores,
                                   const float* __restrict__ blstm,
                                   float* __restrict__ Hbuf) {
    int n = blockIdx.x;                                // entity id
    int tid = threadIdx.x;                             // 64 threads
    float w[KK];
    int   idx[KK];
    float mx = -1e30f;
    #pragma unroll
    for (int k = 0; k < KK; ++k) {
        int sp = spans[n * KK + k];
        bool msk = sp >= 0;
        idx[k] = msk ? sp : 0;
        w[k]   = msk ? scores[idx[k]] : -1e30f;
        mx = fmaxf(mx, w[k]);
    }
    float ssum = 0.0f;
    #pragma unroll
    for (int k = 0; k < KK; ++k) { w[k] = expf(w[k] - mx); ssum += w[k]; }
    float inv = 1.0f / ssum;
    #pragma unroll
    for (int k = 0; k < KK; ++k) w[k] *= inv;
    for (int d = tid; d < H2C; d += blockDim.x) {
        float acc = 0.0f;
        #pragma unroll
        for (int k = 0; k < KK; ++k)
            acc = fmaf(w[k], blstm[(size_t)idx[k] * H2C + d], acc);
        Hbuf[(size_t)n * H2C + d] = acc;
    }
}

// ---------------- K5: state init (Hbuf tail rows, preds in d_out, c state) -------
__global__ void init_state_kernel(const float* __restrict__ Hin,
                                  float* __restrict__ Hbuf,
                                  float* __restrict__ preds,
                                  float* __restrict__ cst) {
    int i = blockIdx.x * blockDim.x + threadIdx.x;
    if (i < NSC * H2C) Hbuf[(size_t)NEC * H2C + i] = Hin[(size_t)NEC * H2C + i];
    if (i < NEC + NSC) { preds[2 * i] = 0.0f; preds[2 * i + 1] = (i < NEC) ? 1.0f : 0.0f; }
    if (i < H2C) cst[i] = 0.0f;
}

// ---------------- K6: sequential DAG-LSTM scan (single workgroup, 1024 threads) ---
__global__ void dag_scan_kernel(const int* __restrict__ T,
                                const int* __restrict__ S,
                                const int* __restrict__ enames,
                                const float* __restrict__ eemb,
                                const float* __restrict__ W_iou,
                                const float* __restrict__ b_iou,
                                const float* __restrict__ Wf,
                                const float* __restrict__ bf,
                                const float* __restrict__ W1,
                                const float* __restrict__ b1,
                                const float* __restrict__ W2,
                                const float* __restrict__ b2,
                                float* __restrict__ Hbuf,
                                float* __restrict__ preds,
                                const float* __restrict__ cst) {
    __shared__ float z[ZDIM];
    __shared__ float iou[IOU3];
    __shared__ float f12[2 * H2C];
    __shared__ float cn[H2C];
    __shared__ float ho[H2C];
    __shared__ float hid1[2 * H2C];
    __shared__ float lg[2];
    __shared__ int   flags[2];
    __shared__ float c[H2C];

    int tid = threadIdx.x;                             // 0..1023
    if (tid < H2C) c[tid] = cst[tid];
    __syncthreads();

    for (int s = 0; s < NSC; ++s) {
        int tgt = T[s];
        int a0  = S[2 * s];
        int a1  = S[2 * s + 1];
        int en  = enames[s];

        if (tid == 0) {
            int valid = (tgt >= NEC) ? 1 : 0;
            int cond  = valid && (preds[2 * a0 + 1] > 0.5f)
                              && (preds[2 * a1 + 1] > 0.5f);
            flags[0] = cond; flags[1] = valid;
        }
        // build z = [e, hv0, hv1]  (1280 > 1024 threads -> strided loop)
        for (int i = tid; i < ZDIM; i += 1024) {
            float v;
            if (i < RR)            v = eemb[(size_t)en * RR + i];
            else if (i < RR + H2C) v = Hbuf[(size_t)a0 * H2C + (i - RR)];
            else                   v = Hbuf[(size_t)a1 * H2C + (i - RR - H2C)];
            z[i] = v;
        }
        __syncthreads();

        // iou = z @ W_iou + b_iou   (1280 x 1536)
        for (int j = tid; j < IOU3; j += 1024) {
            float acc = b_iou[j];
            #pragma unroll 8
            for (int k = 0; k < ZDIM; ++k)
                acc = fmaf(z[k], W_iou[(size_t)k * IOU3 + j], acc);
            iou[j] = acc;
        }
        // f1 (threads 0..511) / f2 (threads 512..1023): [e, hv_w] @ Wf[w] + bf[w]
        {
            int which = tid >> 9;
            int j = tid & (H2C - 1);
            const float* Wfp = Wf + (size_t)which * (RR + H2C) * H2C;
            float acc = bf[which * H2C + j];
            #pragma unroll 8
            for (int k = 0; k < RR; ++k)
                acc = fmaf(z[k], Wfp[(size_t)k * H2C + j], acc);
            #pragma unroll 8
            for (int k = 0; k < H2C; ++k)
                acc = fmaf(z[RR + which * H2C + k], Wfp[(size_t)(RR + k) * H2C + j], acc);
            f12[tid] = sigf(acc);
        }
        __syncthreads();

        // gates -> c_new, h_out   (split: ig=iou[0:512], og=[512:1024], ug=[1024:1536])
        if (tid < H2C) {
            float cc = sigf(iou[tid]) * tanhf(iou[2 * H2C + tid])
                     + f12[tid] * c[tid] + f12[H2C + tid] * c[tid];
            cn[tid] = cc;
            ho[tid] = sigf(iou[H2C + tid]) * tanhf(cc);
        }
        __syncthreads();

        // hid1 = h_out @ W1 + b1   (512 x 1024)
        {
            float acc = b1[tid];
            #pragma unroll 8
            for (int k = 0; k < H2C; ++k)
                acc = fmaf(ho[k], W1[(size_t)k * (2 * H2C) + tid], acc);
            hid1[tid] = acc;
        }
        __syncthreads();

        // logits = hid1 @ W2 + b2   (1024 x 2)
        if (tid < 2) {
            float acc = b2[tid];
            for (int k = 0; k < 2 * H2C; ++k)
                acc = fmaf(hid1[k], W2[2 * k + tid], acc);
            lg[tid] = acc;
        }
        __syncthreads();

        int cond = flags[0], valid = flags[1];
        if (tid == 0 && valid) {
            float r0, r1;
            if (cond) {
                float m  = fmaxf(lg[0], lg[1]);
                float e0 = expf(lg[0] - m), e1 = expf(lg[1] - m);
                float inv = 1.0f / (e0 + e1);
                r0 = e0 * inv; r1 = e1 * inv;
            } else { r0 = 1.0f; r1 = 0.0f; }
            preds[2 * tgt] = r0; preds[2 * tgt + 1] = r1;
        }
        if (cond && tid < H2C) {
            Hbuf[(size_t)tgt * H2C + tid] = ho[tid];
            c[tid] = cn[tid];
        }
        __syncthreads();
    }
}

// ---------------- workspace layout (bytes) ----------------
#define OFF_X16  ((size_t)0)                                   // L*D f16      = 1 MB
#define OFF_WTF  (OFF_X16 + (size_t)LL * DD * 2)               // G4*D f16     = 512 KB
#define OFF_WTB  (OFF_WTF + (size_t)G4 * DD * 2)               // G4*D f16     = 512 KB
#define OFF_XGF  (OFF_WTB + (size_t)G4 * DD * 2)               // L*4H f32     = 8 MB
#define OFF_XGB  (OFF_XGF + (size_t)LL * G4 * 4)               // L*4H f32     = 8 MB
#define OFF_BL   (OFF_XGB + (size_t)LL * G4 * 4)               // L*H2 f32     = 4 MB
#define OFF_SC   (OFF_BL  + (size_t)LL * H2C * 4)              // L    f32
#define OFF_HB   (OFF_SC  + (size_t)LL * 4)                    // (NE+NS)*H2   = 3 MB
#define OFF_CST  (OFF_HB  + (size_t)(NEC + NSC) * H2C * 4)     // H2   f32

extern "C" void kernel_launch(void* const* d_in, const int* in_sizes, int n_in,
                              void* d_out, int out_size, void* d_ws, size_t ws_size,
                              hipStream_t stream) {
    const int*   tokens = (const int*)  d_in[0];
    const int*   spans  = (const int*)  d_in[1];
    const int*   enames = (const int*)  d_in[2];
    const float* Hin    = (const float*)d_in[3];
    /* d_in[4] = A, unused */
    const int*   T      = (const int*)  d_in[5];
    const int*   S      = (const int*)  d_in[6];
    const float* wemb   = (const float*)d_in[7];
    const float* eemb   = (const float*)d_in[8];
    const float* Wa     = (const float*)d_in[9];
    const float* ba     = (const float*)d_in[10];
    const float* WihF   = (const float*)d_in[11];
    const float* WhhF   = (const float*)d_in[12];
    const float* bF     = (const float*)d_in[13];
    const float* WihB   = (const float*)d_in[14];
    const float* WhhB   = (const float*)d_in[15];
    const float* bB     = (const float*)d_in[16];
    const float* W_iou  = (const float*)d_in[17];
    const float* b_iou  = (const float*)d_in[18];
    const float* Wf     = (const float*)d_in[19];
    const float* bf     = (const float*)d_in[20];
    const float* W1     = (const float*)d_in[21];
    const float* b1     = (const float*)d_in[22];
    const float* W2     = (const float*)d_in[23];
    const float* b2     = (const float*)d_in[24];

    float* preds = (float*)d_out;
    char*  ws    = (char*)d_ws;
    _Float16* X16 = (_Float16*)(ws + OFF_X16);
    _Float16* WTF = (_Float16*)(ws + OFF_WTF);
    _Float16* WTB = (_Float16*)(ws + OFF_WTB);
    float* XGF = (float*)(ws + OFF_XGF);
    float* XGB = (float*)(ws + OFF_XGB);
    float* BL  = (float*)(ws + OFF_BL);
    float* SC  = (float*)(ws + OFF_SC);
    float* HB  = (float*)(ws + OFF_HB);
    float* CST = (float*)(ws + OFF_CST);

    // 1) embedding gather -> f16; weight transpose+convert -> f16 (one-time)
    embed_f16_kernel<<<(LL * DD) / 256, 256, 0, stream>>>(tokens, wemb, X16);
    wtrans_f16_kernel<<<(DD * G4) / 256, 256, 0, stream>>>(WihF, WTF, DD, G4);
    wtrans_f16_kernel<<<(DD * G4) / 256, 256, 0, stream>>>(WihB, WTB, DD, G4);

    // 2) hoisted input-gate GEMMs via WMMA:
    //    (L/16)*(G4/64) = 2048 waves, 8 waves/block -> 256 blocks
    gemm_wmma_kernel<<<256, 256, 0, stream>>>(X16, WTF, XGF, LL, G4, DD);
    gemm_wmma_kernel<<<256, 256, 0, stream>>>(X16, WTB, XGB, LL, G4, DD);

    // 3) sequential BiLSTM (two independent single-WGP scans)
    lstm_scan_kernel<<<2, 1024, 0, stream>>>(XGF, XGB, WhhF, WhhB, bF, bB, BL);

    // 4) attention scores + entity pooling
    score_kernel<<<(LL + 255) / 256, 256, 0, stream>>>(BL, Wa, ba, SC);
    entity_pool_kernel<<<NEC, 64, 0, stream>>>(spans, SC, BL, HB);

    // 5) init Hbuf tail rows, preds (in d_out), c state
    init_state_kernel<<<(NSC * H2C + 255) / 256, 256, 0, stream>>>(Hin, HB, preds, CST);

    // 6) sequential DAG-LSTM scan (single persistent workgroup)
    dag_scan_kernel<<<1, 1024, 0, stream>>>(T, S, enames, eemb, W_iou, b_iou,
                                            Wf, bf, W1, b1, W2, b2, HB, preds, CST);
}
